// SelfAttentionHead_45414984188139
// MI455X (gfx1250) — compile-verified, matrix operand order checked by reading, not run on hardware
//
#include <hip/hip_runtime.h>
#include <hip/hip_bf16.h>
#include <math.h>

// Problem constants (match reference)
#define NB     4
#define S_LEN  4096
#define DM     1024
#define DK     64

typedef _Float16 v16h __attribute__((ext_vector_type(16)));
typedef _Float16 v8h  __attribute__((ext_vector_type(8)));
typedef float    v8f  __attribute__((ext_vector_type(8)));
typedef int      v4i  __attribute__((ext_vector_type(4)));
typedef int      v8i  __attribute__((ext_vector_type(8)));

__device__ __forceinline__ v16h cmb(v8h lo, v8h hi) {
  return __builtin_shufflevector(lo, hi, 0,1,2,3,4,5,6,7,8,9,10,11,12,13,14,15);
}

__device__ __forceinline__ v8f wmma_f16(v16h a, v16h b, v8f c) {
  // D = A(16x32 f16) * B(32x16 f16) + C(16x16 f32)
  return __builtin_amdgcn_wmma_f32_16x16x32_f16(false, a, false, b, (short)0, c,
                                                false, false);
}

// ---------------------------------------------------------------------------
// TDM descriptor helpers (CDNA5 ISA).
// Builds a 2-D tile descriptor: tile_d0 elements/row (f16), tile_d1 rows,
// row stride 'stride' elements, LDS padding 'padcode' DWORDs-interval code
// with 4-DWORD pad after each row (conflict-free LDS strides).
// ---------------------------------------------------------------------------
__device__ __forceinline__ v8i tdm_group1(unsigned td0, unsigned td1,
                                          unsigned stride, int interval_code) {
  v8i g1;
  // [15:0] wg_mask=0 | [17:16] data_size=1 (2B) | [20] pad_en
  // [24:22] pad_interval | [31:25] pad_amount=3 (4 DWORDs)
  g1[0] = (1 << 16) | (1 << 20) | (interval_code << 22) | (3 << 25);
  g1[1] = (int)((td0 & 0xFFFFu) << 16);                    // tensor_dim0 lo16
  g1[2] = (int)(((td0 >> 16) & 0xFFFFu) | ((td1 & 0xFFFFu) << 16));
  g1[3] = (int)(((td1 >> 16) & 0xFFFFu) | ((td0 & 0xFFFFu) << 16)); // tile_d0=td0
  g1[4] = (int)(td1 & 0xFFFFu);                            // tile_dim1, tile_dim2=0
  g1[5] = (int)stride;                                     // dim0_stride lo32
  g1[6] = 0;
  g1[7] = 0;
  return g1;
}

__device__ __forceinline__ void tdm_load(unsigned int lds_addr,
                                         const _Float16* gptr, v8i g1) {
  unsigned long long ga = (unsigned long long)(uintptr_t)gptr;
  v4i g0;
  g0[0] = 1;                                   // count=1 (valid user descriptor)
  g0[1] = (int)lds_addr;                       // [63:32] lds_addr (bytes)
  g0[2] = (int)(unsigned int)ga;               // [95:64] global_addr low
  g0[3] = (int)(((unsigned int)(ga >> 32) & 0x01FFFFFFu) | 0x80000000u);
  //          global_addr[56:32] | type=2 in [127:126]
  asm volatile("tensor_load_to_lds %0, %1" :: "s"(g0), "s"(g1) : "memory");
}

// ---------------------------------------------------------------------------
// Prep: transpose + convert the three weight matrices (1024x64 f32) into
// Wt[3][64][1024] f16 so WMMA B-fragments are contiguous along d_model.
// ---------------------------------------------------------------------------
__global__ __launch_bounds__(256) void wprep_kernel(
    const float* __restrict__ Wq, const float* __restrict__ Wk,
    const float* __restrict__ Wv, _Float16* __restrict__ Wt) {
  int i = blockIdx.x * 256 + threadIdx.x;     // 0 .. 3*64*1024-1
  int w   = i >> 16;                          // which matrix
  int rem = i & 65535;
  int c   = rem >> 10;                        // 0..63  (d_k column)
  int r   = rem & 1023;                       // 0..1023 (d_model row)
  const float* W = (w == 0) ? Wq : (w == 1) ? Wk : Wv;
  Wt[i] = (_Float16)W[r * DK + c];
}

// ---------------------------------------------------------------------------
// Fused QKV projection.  Block = 128 threads (4 waves); each wave computes a
// 16-token x 64-col tile of Q, K and V.  Per 32-deep K-step the block's W
// slice (12 KB) is staged in LDS by ONE TDM tensor_load_to_lds (double
// buffered); all 12 B-fragments are preloaded before the 12 WMMAs.
// Outputs: Q,K row-major f16 [B*S][64]; V transposed f16 Vt[B][64][S].
// ---------------------------------------------------------------------------
__global__ __launch_bounds__(128) void qkv_kernel(
    const float* __restrict__ x, const _Float16* __restrict__ Wt,
    const float* __restrict__ bq, const float* __restrict__ bk,
    const float* __restrict__ bv, _Float16* __restrict__ Q,
    _Float16* __restrict__ K, _Float16* __restrict__ Vt) {
  // [buf][3*64 rows][32 data + 8 pad halves]  (TDM pad makes stride 40 halves)
  __shared__ _Float16 ldsW[2][192][40];

  const int wave = threadIdx.x >> 5;
  const int lane = threadIdx.x & 31;
  const int m    = lane & 15;
  const int h    = lane >> 4;
  const int rowbase = blockIdx.x * 64 + wave * 16;
  const int row     = rowbase + m;

  // Wt viewed as [192][1024]; tile 32 x 192; pad_interval=3 (16 DW) -> stride 40
  const v8i g1 = tdm_group1(/*td0=*/32, /*td1=*/192, /*stride=*/1024, 3);
  unsigned int ldsb[2];
  ldsb[0] = (unsigned int)(uintptr_t)&ldsW[0][0][0];
  ldsb[1] = (unsigned int)(uintptr_t)&ldsW[1][0][0];

  v8f zero8 = {};
  v8f qacc[4], kacc[4], vacc[4];
#pragma unroll
  for (int t = 0; t < 4; ++t) { qacc[t] = zero8; kacc[t] = zero8; vacc[t] = zero8; }

  const float* xrow = x + (long)row * DM;

  if (wave == 0) tdm_load(ldsb[0], Wt, g1);   // preload first slice

  int buf = 0;
  for (int k0 = 0; k0 < DM; k0 += 32) {
    if (wave == 0) {
      if (k0 + 32 < DM) {
        tdm_load(ldsb[buf ^ 1], Wt + (k0 + 32), g1);  // prefetch next slice
        __builtin_amdgcn_s_wait_tensorcnt(1);         // current slice done
      } else {
        __builtin_amdgcn_s_wait_tensorcnt(0);
      }
    }
    __syncthreads();

    // A fragment: 16x32 slice of x, f32 -> f16
    v8f xlo = *(const v8f*)(xrow + k0 + 8 * h);
    v8f xhi = *(const v8f*)(xrow + k0 + 16 + 8 * h);
    v16h a = cmb(__builtin_convertvector(xlo, v8h),
                 __builtin_convertvector(xhi, v8h));
    __builtin_prefetch(xrow + ((k0 + 32) & (DM - 1)), 0, 0);

    // preload all 12 B-fragments, then issue the 12 WMMAs
    v16h bw[12];
#pragma unroll
    for (int t = 0; t < 4; ++t) {
      const int col = t * 16 + m;
      const _Float16* wq = &ldsW[buf][col][0];
      const _Float16* wk = &ldsW[buf][64 + col][0];
      const _Float16* wv = &ldsW[buf][128 + col][0];
      bw[t]     = cmb(*(const v8h*)(wq + 8 * h), *(const v8h*)(wq + 16 + 8 * h));
      bw[4 + t] = cmb(*(const v8h*)(wk + 8 * h), *(const v8h*)(wk + 16 + 8 * h));
      bw[8 + t] = cmb(*(const v8h*)(wv + 8 * h), *(const v8h*)(wv + 16 + 8 * h));
    }
#pragma unroll
    for (int t = 0; t < 4; ++t) {
      qacc[t] = wmma_f16(a, bw[t],     qacc[t]);
      kacc[t] = wmma_f16(a, bw[4 + t], kacc[t]);
      vacc[t] = wmma_f16(a, bw[8 + t], vacc[t]);
    }
    __syncthreads();   // all waves done with buf before TDM re-targets it
    buf ^= 1;
  }

  // Store results.  C layout: lane -> col n, element v -> token rowbase+8h+v.
#pragma unroll
  for (int t = 0; t < 4; ++t) {
    const int col = t * 16 + m;
    const float bqs = bq[col], bks = bk[col], bvs = bv[col];
#pragma unroll
    for (int v = 0; v < 8; ++v) {
      const int tok = rowbase + 8 * h + v;
      Q[tok * DK + col] = (_Float16)(qacc[t][v] + bqs);
      K[tok * DK + col] = (_Float16)(kacc[t][v] + bks);
    }
    // V transposed: lane's 8 elements are 8 consecutive tokens -> one 16B store
    const int n0   = rowbase + 8 * h;
    const int bidx = n0 >> 12;           // /S_LEN
    const int s0   = n0 & (S_LEN - 1);
    v8f vv = vacc[t];
#pragma unroll
    for (int v = 0; v < 8; ++v) vv[v] += bvs;
    *(v8h*)(Vt + ((long)(bidx * DK + col) * S_LEN + s0)) =
        __builtin_convertvector(vv, v8h);
  }
}

// ---------------------------------------------------------------------------
// Flash attention (causal).  Block = 256 threads (8 waves) covering 128 query
// rows.  Key loop is block-uniform: K (32x64) and V (64x32) tiles are staged
// in LDS once per block by TDM (double buffered), shared by all 8 waves (8x
// L2-traffic reduction).  Per 32-key block: 4 score WMMAs + 1 row-sum WMMA
// (P x ones, replaces a 32-swizzle reduction) + 4 P*V WMMAs.
// ---------------------------------------------------------------------------
__global__ __launch_bounds__(256) void attn_kernel(
    const _Float16* __restrict__ Q, const _Float16* __restrict__ K,
    const _Float16* __restrict__ Vt, float* __restrict__ out) {
  __shared__ _Float16 ldsK[2][32][72];   // 32 keys x 64 dk, stride 72 (TDM pad)
  __shared__ _Float16 ldsV[2][64][40];   // 64 dk x 32 keys, stride 40 (TDM pad)
  __shared__ _Float16 plds[8][16][40];   // per-wave P re-layout buffer

  const int wave   = threadIdx.x >> 5;
  const int lane   = threadIdx.x & 31;
  const int m      = lane & 15;
  const int h      = lane >> 4;
  const int batch  = blockIdx.y;
  const int qblock = blockIdx.x * 128;
  const int qbase  = qblock + wave * 16;

  const _Float16* Qb = Q + (long)batch * S_LEN * DK;
  const _Float16* Kb = K + (long)batch * S_LEN * DK;
  const _Float16* Vb = Vt + (long)batch * DK * S_LEN;
  float* Ob = out + (long)batch * S_LEN * DK;

  // TDM descriptors: K tile rows are 32 DW + 4 DW pad (code 4); V tile rows
  // are 16 DW + 4 DW pad (code 3).
  const v8i kg1 = tdm_group1(/*td0=*/64, /*td1=*/32, /*stride=*/64, 4);
  const v8i vg1 = tdm_group1(/*td0=*/32, /*td1=*/64, /*stride=*/S_LEN, 3);
  unsigned int ldsKb[2], ldsVb[2];
  ldsKb[0] = (unsigned int)(uintptr_t)&ldsK[0][0][0];
  ldsKb[1] = (unsigned int)(uintptr_t)&ldsK[1][0][0];
  ldsVb[0] = (unsigned int)(uintptr_t)&ldsV[0][0][0];
  ldsVb[1] = (unsigned int)(uintptr_t)&ldsV[1][0][0];

  // Q fragments for the wave's 16 rows; fold in softmax scale 1/sqrt(64)
  v16h aq[2];
#pragma unroll
  for (int c = 0; c < 2; ++c) {
    const _Float16* qp = Qb + (qbase + m) * DK + c * 32;
    v16h a = cmb(*(const v8h*)(qp + 8 * h), *(const v8h*)(qp + 16 + 8 * h));
    aq[c] = a * (_Float16)0.125f;
  }

  v16h ones;
#pragma unroll
  for (int i = 0; i < 16; ++i) ones[i] = (_Float16)1.0f;

  v8f zero8 = {};
  v8f oacc[4], rs_acc = zero8;   // rs_acc accumulates row sums (l) via WMMA
#pragma unroll
  for (int t = 0; t < 4; ++t) oacc[t] = zero8;
  float mi[8];
#pragma unroll
  for (int v = 0; v < 8; ++v) mi[v] = -3.0e38f;

  if (wave == 0) {               // preload first K/V tiles
    tdm_load(ldsKb[0], Kb, kg1);
    tdm_load(ldsVb[0], Vb, vg1);
  }

  const int kbend = qblock + 128;     // block-uniform causal bound
  int buf = 0;
  for (int kb = 0; kb < kbend; kb += 32) {
    if (wave == 0) {
      if (kb + 32 < kbend) {
        tdm_load(ldsKb[buf ^ 1], Kb + (long)(kb + 32) * DK, kg1);
        tdm_load(ldsVb[buf ^ 1], Vb + (kb + 32), vg1);
        __builtin_amdgcn_s_wait_tensorcnt(2);   // current pair complete
      } else {
        __builtin_amdgcn_s_wait_tensorcnt(0);
      }
    }
    __syncthreads();

    // ---- scores: S = Q @ K^T (two 16-key N-tiles, B-frags from LDS) ----
    v8f sc[2];
#pragma unroll
    for (int t = 0; t < 2; ++t) {
      const _Float16* kp = &ldsK[buf][t * 16 + m][0];
      v16h b0 = cmb(*(const v8h*)(kp + 8 * h), *(const v8h*)(kp + 16 + 8 * h));
      v16h b1 = cmb(*(const v8h*)(kp + 32 + 8 * h), *(const v8h*)(kp + 48 + 8 * h));
      v8f z = zero8;
      z = wmma_f16(aq[0], b0, z);
      z = wmma_f16(aq[1], b1, z);
      sc[t] = z;
    }

    // ---- causal mask + online softmax (max only; sums via ones-WMMA) ----
    const int kk0 = kb + m, kk1 = kb + 16 + m;
#pragma unroll
    for (int v = 0; v < 8; ++v) {
      const int rowg = qbase + 8 * h + v;
      float s0v = (kk0 <= rowg) ? sc[0][v] : -3.0e38f;
      float s1v = (kk1 <= rowg) ? sc[1][v] : -3.0e38f;
      float bmax = fmaxf(s0v, s1v);
#pragma unroll
      for (int off = 1; off < 16; off <<= 1)
        bmax = fmaxf(bmax, __shfl_xor(bmax, off, 32));
      const float mnew  = fmaxf(mi[v], bmax);
      const float alpha = __expf(mi[v] - mnew);
      mi[v] = mnew;
      const float p0 = __expf(s0v - mnew);
      const float p1 = __expf(s1v - mnew);
      rs_acc[v] *= alpha;
#pragma unroll
      for (int nt = 0; nt < 4; ++nt) oacc[nt][v] *= alpha;
      // stage P in LDS: row = query (v+8h), col = key-in-block (t*16+m)
      plds[wave][8 * h + v][m]      = (_Float16)p0;
      plds[wave][8 * h + v][16 + m] = (_Float16)p1;
    }
    __builtin_amdgcn_wave_barrier();   // keep DS store->load ordering visible

    // re-read P as an A fragment (16 queries x 32 keys)
    v16h ap = cmb(*(const v8h*)&plds[wave][m][8 * h],
                  *(const v8h*)&plds[wave][m][16 + 8 * h]);
    __builtin_amdgcn_wave_barrier();

    // ---- row sums: rs += P @ ones  (replaces 32-lane-swizzle reduction) ----
    rs_acc = wmma_f16(ap, ones, rs_acc);

    // ---- O += P @ V  (B-frags from TDM-staged, padded LDS) ----
#pragma unroll
    for (int nt = 0; nt < 4; ++nt) {
      const _Float16* vp = &ldsV[buf][nt * 16 + m][0];
      v16h bv = cmb(*(const v8h*)(vp + 8 * h), *(const v8h*)(vp + 16 + 8 * h));
      oacc[nt] = wmma_f16(ap, bv, oacc[nt]);
    }
    __syncthreads();   // all waves done with buf before TDM re-targets it
    buf ^= 1;
  }

  // final normalize + store fp32 output
#pragma unroll
  for (int nt = 0; nt < 4; ++nt) {
#pragma unroll
    for (int v = 0; v < 8; ++v) {
      Ob[(qbase + 8 * h + v) * DK + nt * 16 + m] = oacc[nt][v] / rs_acc[v];
    }
  }
}

// ---------------------------------------------------------------------------
extern "C" void kernel_launch(void* const* d_in, const int* in_sizes, int n_in,
                              void* d_out, int out_size, void* d_ws, size_t ws_size,
                              hipStream_t stream) {
  (void)in_sizes; (void)n_in; (void)out_size; (void)ws_size;
  const float* x  = (const float*)d_in[0];
  // d_in[1] = causal mask, structure known -> not needed
  const float* Wq = (const float*)d_in[2];
  const float* bq = (const float*)d_in[3];
  const float* Wk = (const float*)d_in[4];
  const float* bk = (const float*)d_in[5];
  const float* Wv = (const float*)d_in[6];
  const float* bv = (const float*)d_in[7];
  float* out = (float*)d_out;

  // workspace layout (bytes):
  //   Wt  : 3*64*1024*2 = 393216
  //   Q   : 16384*64*2  = 2097152
  //   K   : 2097152
  //   Vt  : 2097152      (total ~6.6 MB)
  char* ws = (char*)d_ws;
  _Float16* Wt = (_Float16*)ws;
  _Float16* Q  = (_Float16*)(ws + 393216);
  _Float16* K  = (_Float16*)(ws + 393216 + 2097152);
  _Float16* Vt = (_Float16*)(ws + 393216 + 2 * 2097152);

  wprep_kernel<<<768, 256, 0, stream>>>(Wq, Wk, Wv, Wt);
  qkv_kernel<<<(NB * S_LEN) / 64, 128, 0, stream>>>(x, Wt, bq, bk, bv, Q, K, Vt);
  attn_kernel<<<dim3(S_LEN / 128, NB), 256, 0, stream>>>(Q, K, Vt, out);
}
